// RWKV7_18743237279936
// MI455X (gfx1250) — matrix operand order repaired and checked by reading, not compiled, not run on hardware
//
#include <hip/hip_runtime.h>
#include <math.h>

// ---------------------------------------------------------------------------
// Vision-RWKV7 block forward for MI455X (gfx1250, wave32, WMMA).
// Shapes fixed by the harness: B=8, T=4096 (64x64), C=256, H=8, N=32.
// Dense 256x256 projections run on v_wmma_f32_16x16x32_bf16 (bf16 in, f32 acc)
// with weights pre-permuted into fragment order (direct b128 global loads) and
// activations staged in fragment order in double-buffered LDS (b128 ds ops),
// software-pipelined so A-tile fetches and B-fragment loads overlap the WMMAs.
// The wkv7 recurrence runs 1 wave per (batch,head), state in VGPRs, per-step
// vector broadcast via LDS, with global prefetch ahead of the scan.
// ---------------------------------------------------------------------------

#define B_   8
#define T_   4096
#define C_   256
#define H_   8
#define N_   32
#define RESW 64
#define BT_  (B_ * T_)
#define BTC_ ((size_t)BT_ * C_)

typedef __bf16 bf16;
typedef bf16  v16bf __attribute__((ext_vector_type(16)));
typedef float v8f   __attribute__((ext_vector_type(8)));
union V16U { v16bf v; unsigned short u[16]; uint4 q[2]; };

__device__ __forceinline__ unsigned short f2bf(float f) {
  unsigned int u = __float_as_uint(f);
  u += 0x7FFFu + ((u >> 16) & 1u);           // round-to-nearest-even
  return (unsigned short)(u >> 16);
}
__device__ __forceinline__ unsigned int pk2bf(float lo, float hi) {
  return (unsigned int)f2bf(lo) | ((unsigned int)f2bf(hi) << 16);
}
__device__ __forceinline__ float sigmf(float z) { return 1.0f / (1.0f + __expf(-z)); }

// ---------------------------------------------------------------------------
// K0: convert the four 256x256 f32 weights to bf16 *in WMMA B-fragment order*:
//   dst[kt][ntile][lane][e], lane = (n&15) | ((k5>>4)<<4), e = k5&15
// so the GEMM can load each 32-byte B fragment with two uint4 global loads.
// ---------------------------------------------------------------------------
__global__ void k_w2bf(const float* __restrict__ Wr, const float* __restrict__ Wk,
                       const float* __restrict__ Wv, const float* __restrict__ Wo,
                       unsigned short* __restrict__ out) {
  const int i = blockIdx.x * blockDim.x + threadIdx.x;  // 0..65535
  const int k = i >> 8, n = i & 255;
  const int kt = k >> 5, k5 = k & 31;
  const int ntg = n >> 4, nlo = n & 15;
  const int lane = nlo | ((k5 >> 4) << 4);
  const int e = k5 & 15;
  const size_t dst = (((size_t)kt * 16 + ntg) * 32 + lane) * 16 + e;
  out[dst]           = f2bf(Wr[i]);
  out[65536 + dst]   = f2bf(Wk[i]);
  out[131072 + dst]  = f2bf(Wv[i]);
  out[196608 + dst]  = f2bf(Wo[i]);
}

// ---------------------------------------------------------------------------
// K1: per-token quarter-channel 4-direction shift + token-mix interpolation.
// One 256-thread block per token.
// ---------------------------------------------------------------------------
__global__ void __launch_bounds__(256) k_shift_mix(
    const float* __restrict__ X,
    const float* __restrict__ maa_x, const float* __restrict__ maa_rg,
    const float* __restrict__ maa_wa, const float* __restrict__ maa_k,
    const float* __restrict__ maa_v,
    const float* __restrict__ W1,   // (256,112)
    const float* __restrict__ W2,   // (4,28,256)
    float* __restrict__ xrg, float* __restrict__ xwa,
    float* __restrict__ xk,  float* __restrict__ xv) {
  __shared__ float xs[C_], xxs[C_], xms[C_], hid[112];
  const int bt = blockIdx.x, c = threadIdx.x;
  const int b = bt >> 12, t = bt & (T_ - 1);
  const int y = t >> 6, x = t & (RESW - 1);
  const size_t base = (size_t)bt * C_;

  float xc = X[base + c];
  const int q = (c >> 3) & 3;                 // quarter of 32-chan head
  const int sy = y + ((q == 2) ? -1 : (q == 3) ? 1 : 0);
  const int sx = x + ((q == 0) ? -1 : (q == 1) ? 1 : 0);
  float xsrc = 0.0f;
  if (sy >= 0 && sy < RESW && sx >= 0 && sx < RESW)
    xsrc = X[((size_t)b * T_ + (size_t)sy * RESW + sx) * C_ + c];
  const float xx = xsrc - xc;

  xs[c] = xc; xxs[c] = xx; xms[c] = xc + xx * maa_x[c];
  __syncthreads();

  if (c < 112) {
    float s = 0.0f;
    for (int k = 0; k < C_; ++k) s += xms[k] * W1[k * 112 + c];
    hid[c] = tanhf(s);
  }
  __syncthreads();

  float m[4];
#pragma unroll
  for (int l = 0; l < 4; ++l) {
    float s = 0.0f;
    for (int d = 0; d < 28; ++d) s += hid[l * 28 + d] * W2[(l * 28 + d) * C_ + c];
    m[l] = s;
  }
  const float xcv = xs[c], xxv = xxs[c];
  xrg[base + c] = xcv + xxv * (maa_rg[c] + m[0]);
  xwa[base + c] = xcv + xxv * (maa_wa[c] + m[1]);
  xk [base + c] = xcv + xxv * (maa_k[c]  + m[2]);
  xv [base + c] = xcv + xxv * (maa_v[c]  + m[3]);
}

// ---------------------------------------------------------------------------
// K2: C[M,256] = A[M,256] (f32) x W[256,256] (bf16, fragment-permuted), f32.
// Block = 8 waves, 128x128 tile; wave w owns rows [16w,16w+16) x 8 N-tiles.
// Pipelined: A raw data for step kt+1 is global-loaded before the barrier;
// all 8 B fragments are preloaded into distinct registers so the 16
// global_load_b128 issue as one clause and overlap the 8-WMMA chain.
// A-fragment mapping (contiguous k, same order as the K0 B packing):
//   lane = (m&15) | ((k>>4)<<4),  e = k&15
// ---------------------------------------------------------------------------
__global__ void __launch_bounds__(256) k_gemm_bf16(
    const float* __restrict__ A, const unsigned short* __restrict__ PB,
    float* __restrict__ Cm) {
  __shared__ __align__(16) unsigned short AsP[2][128 * 32];  // 2 x 8KB
  const int tid = threadIdx.x, lane = tid & 31, wid = tid >> 5;
  const int mblk = blockIdx.x * 128;
  const int nblk = blockIdx.y * 128;
  const int ntg0 = blockIdx.y * 8;        // global 16-wide N-tile base

  v8f acc[8];
#pragma unroll
  for (int i = 0; i < 8; ++i)
#pragma unroll
    for (int r = 0; r < 8; ++r) acc[i][r] = 0.0f;

  // Per-thread staging assignment: 2 chunks of (row m, 8 consecutive k).
  const int m0_ = (tid + 0)   >> 2, kb0_ = (tid + 0)   & 3;
  const int m1_ = (tid + 256) >> 2, kb1_ = (tid + 256) & 3;
  const float* arow0 = &A[(size_t)(mblk + m0_) * C_ + kb0_ * 8];
  const float* arow1 = &A[(size_t)(mblk + m1_) * C_ + kb1_ * 8];

  float4 pf[2][2];
  pf[0][0] = ((const float4*)arow0)[0];
  pf[0][1] = ((const float4*)arow0)[1];
  pf[1][0] = ((const float4*)arow1)[0];
  pf[1][1] = ((const float4*)arow1)[1];

  for (int kt = 0; kt < 8; ++kt) {
    unsigned short* buf = AsP[kt & 1];
    // Pack + store the prefetched A chunks in fragment order.
#pragma unroll
    for (int i = 0; i < 2; ++i) {
      const int m = (i == 0) ? m0_ : m1_;
      const int kb = (i == 0) ? kb0_ : kb1_;
      uint4 pk;
      pk.x = pk2bf(pf[i][0].x, pf[i][0].y);
      pk.y = pk2bf(pf[i][0].z, pf[i][0].w);
      pk.z = pk2bf(pf[i][1].x, pf[i][1].y);
      pk.w = pk2bf(pf[i][1].z, pf[i][1].w);
      const int lm = (m & 15) | ((kb >> 1) << 4);
      *(uint4*)&buf[((m >> 4) * 32 + lm) * 16 + ((kb & 1) << 3)] = pk;
    }
    // Issue next K-step's A loads before the barrier (hidden by the WMMAs).
    if (kt < 7) {
      const int koff = (kt + 1) * 32;
      pf[0][0] = ((const float4*)(arow0 + koff))[0];
      pf[0][1] = ((const float4*)(arow0 + koff))[1];
      pf[1][0] = ((const float4*)(arow1 + koff))[0];
      pf[1][1] = ((const float4*)(arow1 + koff))[1];
    }
    __syncthreads();

    V16U af;
    {
      const uint4* pa = (const uint4*)&buf[(wid * 32 + lane) * 16];
      af.q[0] = pa[0];
      af.q[1] = pa[1];
    }
    // Preload all 8 B fragments into distinct registers (one load clause).
    V16U bfv[8];
#pragma unroll
    for (int nt = 0; nt < 8; ++nt) {
      const uint4* pb = (const uint4*)
          &PB[(((size_t)kt * 16 + ntg0 + nt) * 32 + lane) * 16];
      bfv[nt].q[0] = pb[0];
      bfv[nt].q[1] = pb[1];
    }
#pragma unroll
    for (int nt = 0; nt < 8; ++nt) {
      acc[nt] = __builtin_amdgcn_wmma_f32_16x16x32_bf16(
          false, af.v, false, bfv[nt].v, (short)0, acc[nt], false, false);
    }
    // No trailing barrier: next iteration writes the other buffer, and a wave
    // can only rewrite this buffer after passing the next barrier, which
    // implies every wave has finished reading it.
  }

  const int nlo = lane & 15, mhi = (lane >> 4) * 8;
#pragma unroll
  for (int nt = 0; nt < 8; ++nt)
#pragma unroll
    for (int r = 0; r < 8; ++r) {
      int row = mblk + wid * 16 + r + mhi;
      int col = nblk + nt * 16 + nlo;
      Cm[(size_t)row * C_ + col] = acc[nt][r];
    }
}

// ---------------------------------------------------------------------------
// K3: low-rank paths + k/v finalization per token (256 threads / token).
// ---------------------------------------------------------------------------
__global__ void __launch_bounds__(256) k_post(
    const float* __restrict__ xrg, const float* __restrict__ xwa,
    const float* __restrict__ xk,  const float* __restrict__ xv,
    const float* __restrict__ v1,
    const float* __restrict__ td,   const float* __restrict__ dw1, const float* __restrict__ dw2,
    const float* __restrict__ aaaaa,const float* __restrict__ aw1, const float* __restrict__ aw2,
    const float* __restrict__ misc_a, const float* __restrict__ maw1, const float* __restrict__ maw2,
    const float* __restrict__ misc_k, const float* __restrict__ mkw1, const float* __restrict__ mkw2,
    const float* __restrict__ misc_v, const float* __restrict__ mvw1, const float* __restrict__ mvw2,
    const float* __restrict__ gw1,  const float* __restrict__ gw2,
    const float* __restrict__ kkw1, const float* __restrict__ kkw2,
    float* __restrict__ wout, float* __restrict__ kio, float* __restrict__ vio,
    float* __restrict__ kkout, float* __restrict__ aout, float* __restrict__ gout) {
  __shared__ float sxrg[C_], sxwa[C_], sxk[C_], sxv[C_];
  __shared__ float hid[128], kkarr[C_], nrm[H_];
  const int bt = blockIdx.x, c = threadIdx.x;
  const size_t base = (size_t)bt * C_;

  const float xrgc = xrg[base + c], xwac = xwa[base + c];
  const float xkc = xk[base + c], xvc = xv[base + c];
  sxrg[c] = xrgc; sxwa[c] = xwac; sxk[c] = xkc; sxv[c] = xvc;
  const float k0 = kio[base + c], v0 = vio[base + c], v1c = v1[base + c];
  __syncthreads();

  // decay: w = -softplus(-(td + tanh(xwa@dw1)@dw2)) - 0.5
  if (c < 64) {
    float s = 0.0f;
    for (int i = 0; i < C_; ++i) s += sxwa[i] * dw1[i * 64 + c];
    hid[c] = tanhf(s);
  }
  __syncthreads();
  float yv = td[c];
  for (int j = 0; j < 64; ++j) yv += hid[j] * dw2[j * C_ + c];
  const float wc = ((yv > 0.0f) ? -log1pf(__expf(-yv)) : (yv - log1pf(__expf(yv)))) - 0.5f;
  wout[base + c] = wc;
  __syncthreads();

  // a = sigmoid(aaaaa + xwa@aw1@aw2)
  if (c < 24) {
    float s = 0.0f;
    for (int i = 0; i < C_; ++i) s += sxwa[i] * aw1[i * 24 + c];
    hid[c] = s;
  }
  __syncthreads();
  float s_a = aaaaa[c];
  for (int j = 0; j < 24; ++j) s_a += hid[j] * aw2[j * C_ + c];
  const float a_c = sigmf(s_a);
  aout[base + c] = a_c;
  __syncthreads();

  // ma = sigmoid(misc_a + xwa@maw1@maw2)
  if (c < 24) {
    float s = 0.0f;
    for (int i = 0; i < C_; ++i) s += sxwa[i] * maw1[i * 24 + c];
    hid[c] = s;
  }
  __syncthreads();
  float s_ma = misc_a[c];
  for (int j = 0; j < 24; ++j) s_ma += hid[j] * maw2[j * C_ + c];
  const float ma_c = sigmf(s_ma);
  __syncthreads();

  // mk = sigmoid(misc_k + xk@mkw1@mkw2)
  if (c < 24) {
    float s = 0.0f;
    for (int i = 0; i < C_; ++i) s += sxk[i] * mkw1[i * 24 + c];
    hid[c] = s;
  }
  __syncthreads();
  float s_mk = misc_k[c];
  for (int j = 0; j < 24; ++j) s_mk += hid[j] * mkw2[j * C_ + c];
  const float mk_c = sigmf(s_mk);
  __syncthreads();

  // v = v0 + (v1 - v0) * sigmoid(misc_v + xv@mvw1@mvw2)
  if (c < 24) {
    float s = 0.0f;
    for (int i = 0; i < C_; ++i) s += sxv[i] * mvw1[i * 24 + c];
    hid[c] = s;
  }
  __syncthreads();
  float s_mv = misc_v[c];
  for (int j = 0; j < 24; ++j) s_mv += hid[j] * mvw2[j * C_ + c];
  vio[base + c] = v0 + (v1c - v0) * sigmf(s_mv);
  __syncthreads();

  // g = sigmoid(xrg@gw1) @ gw2
  if (c < 120) {
    float s = 0.0f;
    for (int i = 0; i < C_; ++i) s += sxrg[i] * gw1[i * 120 + c];
    hid[c] = sigmf(s);
  }
  __syncthreads();
  float s_g = 0.0f;
  for (int j = 0; j < 120; ++j) s_g += hid[j] * gw2[j * C_ + c];
  gout[base + c] = s_g;
  __syncthreads();

  // kk = normalize_per_head(k0 + tanh(xk@kkw1)@kkw2)
  if (c < 24) {
    float s = 0.0f;
    for (int i = 0; i < C_; ++i) s += sxk[i] * kkw1[i * 24 + c];
    hid[c] = tanhf(s);
  }
  __syncthreads();
  float s_kk = k0;
  for (int j = 0; j < 24; ++j) s_kk += hid[j] * kkw2[j * C_ + c];
  kkarr[c] = s_kk;
  __syncthreads();
  if (c < H_) {
    float s = 0.0f;
    for (int n = 0; n < N_; ++n) { float t = kkarr[c * N_ + n]; s += t * t; }
    nrm[c] = fmaxf(sqrtf(s), 1e-12f);
  }
  __syncthreads();
  kkout[base + c] = kkarr[c] / nrm[c >> 5];

  // k = (k0*ma + k0*a*(1-ma)) * exp(min(w*mk, 0))
  const float kmix = k0 * ma_c + k0 * a_c * (1.0f - ma_c);
  kio[base + c] = kmix * __expf(fminf(wc * mk_c, 0.0f));
}

// ---------------------------------------------------------------------------
// K4: wkv7 scan. 8 blocks (batch) x 8 waves (head); lane = value row i.
// S[i][j] lives in 32 VGPRs per lane; per-step j-vectors broadcast via LDS
// (one b32 + one b128 load per j). a_t = -kk, b_t = kk*a, wexp = exp(w).
// ---------------------------------------------------------------------------
__global__ void __launch_bounds__(256) k_wkv7(
    const float* __restrict__ r, const float* __restrict__ w,
    const float* __restrict__ k, const float* __restrict__ v,
    const float* __restrict__ kk, const float* __restrict__ a,
    float* __restrict__ o) {
  __shared__ float  aaS[H_][N_];
  __shared__ float4 pkS[H_][N_];
  const int b = blockIdx.x;
  const int h = threadIdx.x >> 5, lane = threadIdx.x & 31;

  float S[32];
#pragma unroll
  for (int j = 0; j < 32; ++j) S[j] = 0.0f;

  size_t base = (size_t)b * T_ * C_ + h * N_ + lane;
  for (int t = 0; t < T_; ++t, base += C_) {
    const float rr = r[base];
    const float ww = __expf(w[base]);
    const float kv_ = k[base];
    const float vv_ = v[base];
    const float kkv = kk[base];
    const float av_ = a[base];
    if (t + 16 < T_) {
      __builtin_prefetch(&r[base + 16 * C_], 0, 0);
      __builtin_prefetch(&k[base + 16 * C_], 0, 0);
      __builtin_prefetch(&v[base + 16 * C_], 0, 0);
    }
    aaS[h][lane] = -kkv;                       // a_t (delta-rule "erase" vec)
    pkS[h][lane] = make_float4(ww, kkv * av_, kv_, rr);

    float sa = 0.0f;
#pragma unroll
    for (int j = 0; j < 32; ++j) sa += S[j] * aaS[h][j];

    float out = 0.0f;
#pragma unroll
    for (int j = 0; j < 32; ++j) {
      const float4 p = pkS[h][j];              // (wexp, b, k, r)
      S[j] = S[j] * p.x + sa * p.y + vv_ * p.z;
      out += S[j] * p.w;
    }
    o[base] = out;
  }
}

// ---------------------------------------------------------------------------
// K5: per-head groupnorm + (r*k*faaaa).sum * v residual + gate multiply.
// ---------------------------------------------------------------------------
__global__ void __launch_bounds__(256) k_gn(
    const float* __restrict__ r, const float* __restrict__ k,
    const float* __restrict__ v, const float* __restrict__ g,
    const float* __restrict__ faaaa,
    const float* __restrict__ gamma, const float* __restrict__ beta,
    float* __restrict__ o) {
  __shared__ float so[C_], sr[C_], sk[C_];
  __shared__ float mu[H_], rsig[H_], rkh[H_];
  const int bt = blockIdx.x, c = threadIdx.x;
  const size_t base = (size_t)bt * C_;
  so[c] = o[base + c]; sr[c] = r[base + c]; sk[c] = k[base + c];
  __syncthreads();
  if (c < H_) {
    float m = 0.0f;
    for (int n = 0; n < N_; ++n) m += so[c * N_ + n];
    m *= (1.0f / N_);
    float vv = 0.0f;
    for (int n = 0; n < N_; ++n) { float d = so[c * N_ + n] - m; vv += d * d; }
    vv *= (1.0f / N_);
    mu[c] = m; rsig[c] = rsqrtf(vv + 6.4e-5f);
    float s = 0.0f;
    for (int n = 0; n < N_; ++n) s += sr[c * N_ + n] * sk[c * N_ + n] * faaaa[c * N_ + n];
    rkh[c] = s;
  }
  __syncthreads();
  const int hh = c >> 5;
  const float on = (so[c] - mu[hh]) * rsig[hh] * gamma[c] + beta[c];
  const float o2 = on + rkh[hh] * v[base + c];
  o[base + c] = o2 * g[base + c];
}

// ---------------------------------------------------------------------------
// Launch. Input order = setup_inputs() dict order (params flattened in dict
// insertion order after x, v1, res_h, res_w, scan).
// ---------------------------------------------------------------------------
extern "C" void kernel_launch(void* const* d_in, const int* in_sizes, int n_in,
                              void* d_out, int out_size, void* d_ws, size_t ws_size,
                              hipStream_t stream) {
  const float* X  = (const float*)d_in[0];
  const float* V1 = (const float*)d_in[1];
  // d_in[2..4] = res_h, res_w, scan (fixed 64/64/0 by the harness setup)
  const float* maa_x  = (const float*)d_in[5];
  const float* maa_rg = (const float*)d_in[6];
  const float* maa_wa = (const float*)d_in[7];
  const float* maa_k  = (const float*)d_in[8];
  const float* maa_v  = (const float*)d_in[9];
  const float* tdecay = (const float*)d_in[10];
  const float* faaaa  = (const float*)d_in[11];
  const float* aaaaa  = (const float*)d_in[12];
  const float* W1     = (const float*)d_in[13];
  const float* W2     = (const float*)d_in[14];
  const float* dw1    = (const float*)d_in[15];
  const float* dw2    = (const float*)d_in[16];
  const float* aw1    = (const float*)d_in[17];
  const float* aw2    = (const float*)d_in[18];
  const float* kkw1   = (const float*)d_in[19];
  const float* kkw2   = (const float*)d_in[20];
  const float* gw1    = (const float*)d_in[21];
  const float* gw2    = (const float*)d_in[22];
  const float* maw1   = (const float*)d_in[23];
  const float* maw2   = (const float*)d_in[24];
  const float* misc_a = (const float*)d_in[25];
  const float* mkw1   = (const float*)d_in[26];
  const float* mkw2   = (const float*)d_in[27];
  const float* misc_k = (const float*)d_in[28];
  const float* mvw1   = (const float*)d_in[29];
  const float* mvw2   = (const float*)d_in[30];
  const float* misc_v = (const float*)d_in[31];
  const float* Wr     = (const float*)d_in[32];
  const float* Wk     = (const float*)d_in[33];
  const float* Wv     = (const float*)d_in[34];
  const float* Wo     = (const float*)d_in[35];
  const float* gamma  = (const float*)d_in[36];
  const float* beta   = (const float*)d_in[37];

  float* ws  = (float*)d_ws;                 // needs ~12*BTC*4 + 512KB bytes
  float* xrg = ws + 0 * BTC_;
  float* xwa = ws + 1 * BTC_;
  float* xk  = ws + 2 * BTC_;
  float* xv  = ws + 3 * BTC_;
  float* rb  = ws + 4 * BTC_;
  float* wb  = ws + 5 * BTC_;
  float* kb  = ws + 6 * BTC_;
  float* vb  = ws + 7 * BTC_;
  float* kkb = ws + 8 * BTC_;
  float* ab  = ws + 9 * BTC_;
  float* gb  = ws + 10 * BTC_;
  float* ob  = ws + 11 * BTC_;
  unsigned short* wbf = (unsigned short*)(ws + 12 * BTC_);  // 4 x 65536 bf16

  k_w2bf<<<256, 256, 0, stream>>>(Wr, Wk, Wv, Wo, wbf);

  k_shift_mix<<<BT_, 256, 0, stream>>>(X, maa_x, maa_rg, maa_wa, maa_k, maa_v,
                                       W1, W2, xrg, xwa, xk, xv);

  dim3 gg(BT_ / 128, 2);
  k_gemm_bf16<<<gg, 256, 0, stream>>>(xrg, wbf + 0,      rb);
  k_gemm_bf16<<<gg, 256, 0, stream>>>(xk,  wbf + 65536,  kb);
  k_gemm_bf16<<<gg, 256, 0, stream>>>(xv,  wbf + 131072, vb);

  k_post<<<BT_, 256, 0, stream>>>(xrg, xwa, xk, xv, V1,
                                  tdecay, dw1, dw2,
                                  aaaaa, aw1, aw2,
                                  misc_a, maw1, maw2,
                                  misc_k, mkw1, mkw2,
                                  misc_v, mvw1, mvw2,
                                  gw1, gw2, kkw1, kkw2,
                                  wb, kb, vb, kkb, ab, gb);

  k_wkv7<<<B_, 256, 0, stream>>>(rb, wb, kb, vb, kkb, ab, ob);

  k_gn<<<BT_, 256, 0, stream>>>(rb, kb, vb, gb, faaaa, gamma, beta, ob);

  k_gemm_bf16<<<gg, 256, 0, stream>>>(ob, wbf + 196608, (float*)d_out);
}
